// Attention_21698174779984
// MI455X (gfx1250) — compile-verified, matrix operand order checked by reading, not run on hardware
//
#include <hip/hip_runtime.h>
#include <hip/hip_bf16.h>

typedef __attribute__((ext_vector_type(16))) _Float16 v16h;
typedef __attribute__((ext_vector_type(8)))  _Float16 v8h;
typedef __attribute__((ext_vector_type(4)))  _Float16 v4h;
typedef __attribute__((ext_vector_type(8)))  float    v8f;

#define N_HEADS 8
#define SEQ     2048
#define DIM     128
#define QT      32            // queries per workgroup
#define NWAVES  8
#define BLOCK   (NWAVES * 32)

// XOR swizzle for the LDS score matrix: element (m, k) lives at
//   m*SEQ + (k ^ SWZ(m)),  SWZ(m) = (m & 15) << 3   (multiples of 8 halfs)
// -> keeps 16B-aligned 8-half runs intact, spreads the 16 rows of a WMMA
//    A-operand load across 16 distinct bank offsets (conflict-free), and
//    separates the +8-row half-wave of phase-1 stores onto different banks.
__device__ __forceinline__ int swz_of_row(int m) { return (m & 15) << 3; }

// ---------------------------------------------------------------------------
// WMMA wrapper: D = A(16x32 f16) * B(32x16 f16) + C(16x16 f32)
// ---------------------------------------------------------------------------
__device__ __forceinline__ v8f wmma16(v16h a, v16h b, v8f c) {
  return __builtin_amdgcn_wmma_f32_16x16x32_f16(
      /*neg_a=*/false, a, /*neg_b=*/false, b,
      /*c_mod=*/(short)0, c, /*reuse_a=*/false, /*reuse_b=*/false);
}

// ---------------------------------------------------------------------------
// 16-bit WMMA operand (lane -> row(M/N), halfs -> K) from a row-major f16
// tile in GLOBAL memory. K-packing per ISA 7.12.2:
//   lane<16 : halfs 0..7 -> K 0..7,  halfs 8..15 -> K 16..23
//   lane>=16: halfs 0..7 -> K 8..15, halfs 8..15 -> K 24..31
// ---------------------------------------------------------------------------
__device__ __forceinline__ v16h load_op_f16(const _Float16* base,
                                            int rowStride, int col0, int lane) {
  const int r  = lane & 15;
  const int hi = lane >> 4;
  const _Float16* p = base + (size_t)r * rowStride + col0 + hi * 8;
  v8h lo = *(const v8h*)(p);
  v8h hv = *(const v8h*)(p + 16);
  v16h o;
  o[0]  = lo[0]; o[1]  = lo[1]; o[2]  = lo[2]; o[3]  = lo[3];
  o[4]  = lo[4]; o[5]  = lo[5]; o[6]  = lo[6]; o[7]  = lo[7];
  o[8]  = hv[0]; o[9]  = hv[1]; o[10] = hv[2]; o[11] = hv[3];
  o[12] = hv[4]; o[13] = hv[5]; o[14] = hv[6]; o[15] = hv[7];
  return o;
}

// Same operand from the XOR-swizzled LDS score matrix (row stride = SEQ).
__device__ __forceinline__ v16h load_op_lds_sw(const _Float16* base,
                                               int col0, int lane) {
  const int r   = lane & 15;
  const int hi  = lane >> 4;
  const int swz = r << 3;
  const _Float16* prow = base + (size_t)r * SEQ;
  v8h lo = *(const v8h*)(prow + ((col0 + hi * 8)      ^ swz));
  v8h hv = *(const v8h*)(prow + ((col0 + hi * 8 + 16) ^ swz));
  v16h o;
  o[0]  = lo[0]; o[1]  = lo[1]; o[2]  = lo[2]; o[3]  = lo[3];
  o[4]  = lo[4]; o[5]  = lo[5]; o[6]  = lo[6]; o[7]  = lo[7];
  o[8]  = hv[0]; o[9]  = hv[1]; o[10] = hv[2]; o[11] = hv[3];
  o[12] = hv[4]; o[13] = hv[5]; o[14] = hv[6]; o[15] = hv[7];
  return o;
}

// WMMA operand from row-major f32 in global memory (fallback path).
__device__ __forceinline__ v16h load_op_f32(const float* __restrict__ base,
                                            int rowStride, int col0, int lane) {
  const int r  = lane & 15;
  const int hi = lane >> 4;
  const float* p = base + (size_t)r * rowStride + col0 + hi * 8;
  float4 a0 = *(const float4*)(p + 0);
  float4 a1 = *(const float4*)(p + 4);
  float4 b0 = *(const float4*)(p + 16);
  float4 b1 = *(const float4*)(p + 20);
  v16h o;
  o[0]  = (_Float16)a0.x; o[1]  = (_Float16)a0.y; o[2]  = (_Float16)a0.z; o[3]  = (_Float16)a0.w;
  o[4]  = (_Float16)a1.x; o[5]  = (_Float16)a1.y; o[6]  = (_Float16)a1.z; o[7]  = (_Float16)a1.w;
  o[8]  = (_Float16)b0.x; o[9]  = (_Float16)b0.y; o[10] = (_Float16)b0.z; o[11] = (_Float16)b0.w;
  o[12] = (_Float16)b1.x; o[13] = (_Float16)b1.y; o[14] = (_Float16)b1.z; o[15] = (_Float16)b1.w;
  return o;
}

// ---------------------------------------------------------------------------
// Prep kernel 1: f32 -> f16 elementwise (Q, K), vectorized 4-wide.
// ---------------------------------------------------------------------------
__global__ __launch_bounds__(256)
void cvt_f32_to_f16(const float* __restrict__ in, _Float16* __restrict__ out, int n4) {
  int i = blockIdx.x * 256 + threadIdx.x;
  if (i < n4) {
    float4 x = ((const float4*)in)[i];
    v4h y;
    y[0] = (_Float16)x.x; y[1] = (_Float16)x.y; y[2] = (_Float16)x.z; y[3] = (_Float16)x.w;
    ((v4h*)out)[i] = y;
  }
}

// ---------------------------------------------------------------------------
// Prep kernel 2: V [h][key][d] f32 -> Vt [h][d][key] f16, LDS-tiled 32x32.
// ---------------------------------------------------------------------------
__global__ __launch_bounds__(256)
void transpose_v_f16(const float* __restrict__ v, _Float16* __restrict__ vt) {
  __shared__ _Float16 tile[32][33];
  const int k0 = blockIdx.x * 32;
  const int n0 = blockIdx.y * 32;
  const int h  = blockIdx.z;
  const int tn = threadIdx.x & 31;
  const int tk = threadIdx.x >> 5;
  const float* src = v + (size_t)h * SEQ * DIM;
#pragma unroll
  for (int i = 0; i < 4; ++i) {
    const int kk = tk + i * 8;
    tile[kk][tn] = (_Float16)src[(size_t)(k0 + kk) * DIM + n0 + tn];
  }
  __syncthreads();
  _Float16* dst = vt + (size_t)h * DIM * SEQ;
#pragma unroll
  for (int i = 0; i < 4; ++i) {
    const int nn = tk + i * 8;
    dst[(size_t)(n0 + nn) * SEQ + k0 + tn] = tile[tn][nn];
  }
}

// ---------------------------------------------------------------------------
// Shared LN + softmax passes over the swizzled LDS score strip (v8h vector).
// ---------------------------------------------------------------------------
__device__ __forceinline__ void ln_softmax_passes(
    _Float16* sS, const float* sG, const float* sBe,
    float* sumP, float* sqP, float* maxP, float* zP,
    float* rowMu, float* rowRstd, float* rowMaxF, float* rowZinv, int tid) {
  const int row   = tid >> 3;
  const int part  = tid & 7;
  const int kbase = part * (SEQ / 8);
  const int swz   = swz_of_row(row);
  _Float16* rowp  = sS + (size_t)row * SEQ;

  // ---- Pass A: row mean / var ----
  {
    float s = 0.f, s2 = 0.f;
    for (int j = 0; j < 32; ++j) {
      const int c8  = (j + tid) & 31;
      const int off = (kbase + c8 * 8) ^ swz;
      v8h x8 = *(const v8h*)(rowp + off);
#pragma unroll
      for (int e = 0; e < 8; ++e) { const float x = (float)x8[e]; s += x; s2 += x * x; }
    }
    sumP[row * 8 + part] = s;
    sqP[row * 8 + part]  = s2;
  }
  __syncthreads();
  if (tid < QT) {
    float s = 0.f, s2 = 0.f;
#pragma unroll
    for (int p2 = 0; p2 < 8; ++p2) { s += sumP[tid * 8 + p2]; s2 += sqP[tid * 8 + p2]; }
    const float mu  = s * (1.0f / SEQ);
    const float var = s2 * (1.0f / SEQ) - mu * mu;
    rowMu[tid]   = mu;
    rowRstd[tid] = rsqrtf(var + 1e-5f);
  }
  __syncthreads();

  // ---- Pass B: s_hat = (s-mu)*rstd*g + b (in place) + row max ----
  {
    const float mu = rowMu[row], rs = rowRstd[row];
    float mx = -3.0e38f;
    for (int j = 0; j < 32; ++j) {
      const int c8      = (j + tid) & 31;
      const int off_log = kbase + c8 * 8;        // logical key index (gamma/beta)
      const int off_s   = off_log ^ swz;         // swizzled LDS offset
      v8h x8 = *(const v8h*)(rowp + off_s);
      float4 g0 = *(const float4*)(sG + off_log);
      float4 g1 = *(const float4*)(sG + off_log + 4);
      float4 b0 = *(const float4*)(sBe + off_log);
      float4 b1 = *(const float4*)(sBe + off_log + 4);
      float g[8] = {g0.x, g0.y, g0.z, g0.w, g1.x, g1.y, g1.z, g1.w};
      float b[8] = {b0.x, b0.y, b0.z, b0.w, b1.x, b1.y, b1.z, b1.w};
      v8h y8;
#pragma unroll
      for (int e = 0; e < 8; ++e) {
        const float sh = ((float)x8[e] - mu) * rs * g[e] + b[e];
        y8[e] = (_Float16)sh;
        mx = fmaxf(mx, sh);
      }
      *(v8h*)(rowp + off_s) = y8;
    }
    maxP[row * 8 + part] = mx;
  }
  __syncthreads();
  if (tid < QT) {
    float mx = maxP[tid * 8];
#pragma unroll
    for (int p2 = 1; p2 < 8; ++p2) mx = fmaxf(mx, maxP[tid * 8 + p2]);
    rowMaxF[tid] = mx;
  }
  __syncthreads();

  // ---- Pass C: p = exp(s_hat - max) (in place) + row Z ----
  {
    const float mx = rowMaxF[row];
    float z = 0.f;
    for (int j = 0; j < 32; ++j) {
      const int c8  = (j + tid) & 31;
      const int off = (kbase + c8 * 8) ^ swz;
      v8h x8 = *(const v8h*)(rowp + off);
      v8h y8;
#pragma unroll
      for (int e = 0; e < 8; ++e) {
        const float p = __expf((float)x8[e] - mx);
        y8[e] = (_Float16)p;
        z += p;
      }
      *(v8h*)(rowp + off) = y8;
    }
    zP[row * 8 + part] = z;
  }
  __syncthreads();
  if (tid < QT) {
    float z = 0.f;
#pragma unroll
    for (int p2 = 0; p2 < 8; ++p2) z += zP[tid * 8 + p2];
    rowZinv[tid] = 1.0f / z;
  }
  __syncthreads();
}

#define SMEM_DECLS                                                         \
  extern __shared__ char smem[];                                           \
  _Float16* sS   = (_Float16*)smem;                                        \
  float*    sG   = (float*)(smem + (size_t)QT * SEQ * 2);                  \
  float*    sBe  = sG + SEQ;                                               \
  float*    sumP = sBe + SEQ;                                              \
  float*    sqP  = sumP + QT * 8;                                          \
  float*    maxP = sqP + QT * 8;                                           \
  float*    zP   = maxP + QT * 8;                                          \
  float*    rowMu   = zP + QT * 8;                                         \
  float*    rowRstd = rowMu + QT;                                          \
  float*    rowMaxF = rowRstd + QT;                                        \
  float*    rowZinv = rowMaxF + QT;

// Store one WMMA accumulator pair into the swizzled score matrix.
__device__ __forceinline__ void store_scores_sw(_Float16* sS, v8f acc0, v8f acc1,
                                                int kk0, int lane, float scale) {
  const int colL = kk0 + (lane & 15);
  const int hi   = lane >> 4;
#pragma unroll
  for (int r = 0; r < 8; ++r) {
    const int m0 = r + 8 * hi;
    sS[(size_t)m0 * SEQ        + (colL ^ swz_of_row(m0))] = (_Float16)(acc0[r] * scale);
    sS[(size_t)(16 + m0) * SEQ + (colL ^ swz_of_row(m0))] = (_Float16)(acc1[r] * scale);
  }
}

// ---------------------------------------------------------------------------
// FAST PATH: Q16/K16 natural f16, Vt16 transposed f16 (from d_ws).
// Four independent WMMA accumulation chains per wave in both GEMM loops so
// dependent WMMAs are >=4 instructions apart (hides the f16 WMMA RAW hazard).
// ---------------------------------------------------------------------------
__global__ __launch_bounds__(BLOCK)
void attn_lnsm_f16(const _Float16* __restrict__ q16, const _Float16* __restrict__ k16,
                   const _Float16* __restrict__ vt16, const float* __restrict__ gamma,
                   const float* __restrict__ beta, float* __restrict__ out) {
  SMEM_DECLS
  const int tid  = threadIdx.x;
  const int lane = tid & 31;
  const int wid  = tid >> 5;
  const int h    = blockIdx.y;
  const int q0   = blockIdx.x * QT;

  const _Float16* Qh = q16 + (size_t)h * SEQ * DIM;
  const _Float16* Kh = k16 + (size_t)h * SEQ * DIM;

  for (int i = tid; i < SEQ; i += BLOCK) { sG[i] = gamma[i]; sBe[i] = beta[i]; }

  // ---------------- Phase 1: S = scale * Q K^T ----------------
  v16h aQ[2][4];
#pragma unroll
  for (int t = 0; t < 2; ++t)
#pragma unroll
    for (int c = 0; c < 4; ++c)
      aQ[t][c] = load_op_f16(Qh + (size_t)(q0 + 16 * t) * DIM, DIM, c * 32, lane);

  const float scale = 0.088388347648318447f; // 1/sqrt(128)
  // Two key tiles per iteration -> 4 independent accumulator chains.
  for (int i = 0; i < (SEQ / 16) / (2 * NWAVES); ++i) {
    const int ktA = wid + 2 * NWAVES * i;
    const int kkA = ktA * 16;
    const int kkB = kkA + 16 * NWAVES;
    v8f a0 = {}, a1 = {}, b0 = {}, b1 = {};
#pragma unroll
    for (int c = 0; c < 4; ++c) {
      v16h bKA = load_op_f16(Kh + (size_t)kkA * DIM, DIM, c * 32, lane);
      v16h bKB = load_op_f16(Kh + (size_t)kkB * DIM, DIM, c * 32, lane);
      a0 = wmma16(aQ[0][c], bKA, a0);
      a1 = wmma16(aQ[1][c], bKA, a1);
      b0 = wmma16(aQ[0][c], bKB, b0);
      b1 = wmma16(aQ[1][c], bKB, b1);
    }
    store_scores_sw(sS, a0, a1, kkA, lane, scale);
    store_scores_sw(sS, b0, b1, kkB, lane, scale);
  }
  __syncthreads();

  ln_softmax_passes(sS, sG, sBe, sumP, sqP, maxP, zP,
                    rowMu, rowRstd, rowMaxF, rowZinv, tid);

  // ---------------- Phase 3: O = (P V) / Z ----------------
  {
    const int n0 = wid * 16;
    const int hi = lane >> 4;
    const int nc = n0 + (lane & 15);
    const _Float16* Vtn = vt16 + ((size_t)h * DIM + n0) * SEQ;
    // Even/odd chunk partial accumulators -> 4 independent chains.
    v8f e0 = {}, e1 = {}, o0 = {}, o1 = {};
    for (int kc = 0; kc < SEQ; kc += 64) {
      v16h A0a = load_op_lds_sw(sS,                    kc, lane);
      v16h A1a = load_op_lds_sw(sS + (size_t)16 * SEQ, kc, lane);
      v16h Ba  = load_op_f16(Vtn, SEQ, kc, lane);
      v16h A0b = load_op_lds_sw(sS,                    kc + 32, lane);
      v16h A1b = load_op_lds_sw(sS + (size_t)16 * SEQ, kc + 32, lane);
      v16h Bb  = load_op_f16(Vtn, SEQ, kc + 32, lane);
      e0 = wmma16(A0a, Ba, e0);
      e1 = wmma16(A1a, Ba, e1);
      o0 = wmma16(A0b, Bb, o0);
      o1 = wmma16(A1b, Bb, o1);
    }
    float* Oh = out + ((size_t)h * SEQ + q0) * DIM;
#pragma unroll
    for (int r = 0; r < 8; ++r) {
      const int m0 = r + 8 * hi;
      Oh[(size_t)m0 * DIM + nc]        = (e0[r] + o0[r]) * rowZinv[m0];
      Oh[(size_t)(16 + m0) * DIM + nc] = (e1[r] + o1[r]) * rowZinv[16 + m0];
    }
  }
}

// ---------------------------------------------------------------------------
// FALLBACK: straight-from-f32 path (no workspace required).
// ---------------------------------------------------------------------------
__global__ __launch_bounds__(BLOCK)
void attn_lnsm_f32(const float* __restrict__ q, const float* __restrict__ k,
                   const float* __restrict__ v, const float* __restrict__ gamma,
                   const float* __restrict__ beta, float* __restrict__ out) {
  SMEM_DECLS
  const int tid  = threadIdx.x;
  const int lane = tid & 31;
  const int wid  = tid >> 5;
  const int h    = blockIdx.y;
  const int q0   = blockIdx.x * QT;

  const float* Qh = q + (size_t)h * SEQ * DIM;
  const float* Kh = k + (size_t)h * SEQ * DIM;
  const float* Vh = v + (size_t)h * SEQ * DIM;

  for (int i = tid; i < SEQ; i += BLOCK) { sG[i] = gamma[i]; sBe[i] = beta[i]; }

  v16h aQ[2][4];
#pragma unroll
  for (int t = 0; t < 2; ++t)
#pragma unroll
    for (int c = 0; c < 4; ++c)
      aQ[t][c] = load_op_f32(Qh + (size_t)(q0 + 16 * t) * DIM, DIM, c * 32, lane);

  const float scale = 0.088388347648318447f;
  for (int kt = wid; kt < SEQ / 16; kt += NWAVES) {
    const int kk0 = kt * 16;
    v8f acc0 = {}; v8f acc1 = {};
#pragma unroll
    for (int c = 0; c < 4; ++c) {
      v16h bK = load_op_f32(Kh + (size_t)kk0 * DIM, DIM, c * 32, lane);
      acc0 = wmma16(aQ[0][c], bK, acc0);
      acc1 = wmma16(aQ[1][c], bK, acc1);
    }
    store_scores_sw(sS, acc0, acc1, kk0, lane, scale);
  }
  __syncthreads();

  ln_softmax_passes(sS, sG, sBe, sumP, sqP, maxP, zP,
                    rowMu, rowRstd, rowMaxF, rowZinv, tid);

  {
    const int n0 = wid * 16;
    const int hi = lane >> 4;
    const int nc = n0 + (lane & 15);
    v8f acc0 = {}; v8f acc1 = {};
    for (int kc = 0; kc < SEQ; kc += 32) {
      v16h A0 = load_op_lds_sw(sS,                    kc, lane);
      v16h A1 = load_op_lds_sw(sS + (size_t)16 * SEQ, kc, lane);
      v16h B;
#pragma unroll
      for (int j = 0; j < 8; ++j) {
        B[j]     = (_Float16)Vh[(size_t)(kc + hi * 8 + j) * DIM + nc];
        B[8 + j] = (_Float16)Vh[(size_t)(kc + 16 + hi * 8 + j) * DIM + nc];
      }
      acc0 = wmma16(A0, B, acc0);
      acc1 = wmma16(A1, B, acc1);
    }
    float* Oh = out + ((size_t)h * SEQ + q0) * DIM;
#pragma unroll
    for (int r = 0; r < 8; ++r) {
      const int m0 = r + 8 * hi;
      Oh[(size_t)m0 * DIM + nc]        = acc0[r] * rowZinv[m0];
      Oh[(size_t)(16 + m0) * DIM + nc] = acc1[r] * rowZinv[16 + m0];
    }
  }
}

extern "C" void kernel_launch(void* const* d_in, const int* in_sizes, int n_in,
                              void* d_out, int out_size, void* d_ws, size_t ws_size,
                              hipStream_t stream) {
  const float* q     = (const float*)d_in[0];
  const float* k     = (const float*)d_in[1];
  const float* v     = (const float*)d_in[2];
  const float* gamma = (const float*)d_in[3];
  const float* beta  = (const float*)d_in[4];
  float* out = (float*)d_out;

  const size_t lds_bytes = (size_t)QT * SEQ * 2
                         + (size_t)SEQ * 4 * 2
                         + (size_t)QT * 8 * 4 * 4
                         + (size_t)QT * 4 * 4;       // = 152,064 B

  dim3 grid(SEQ / QT, N_HEADS);   // 64 x 8 = 512 workgroups
  dim3 block(BLOCK);              // 256 threads = 8 wave32

  const size_t elems    = (size_t)N_HEADS * SEQ * DIM;        // 2,097,152
  const size_t f16bytes = elems * 2;                          // 4 MB each
  const size_t ws_need  = 3 * f16bytes;                       // 12 MB

  if (ws_size >= ws_need) {
    _Float16* q16  = (_Float16*)d_ws;
    _Float16* k16  = (_Float16*)((char*)d_ws + f16bytes);
    _Float16* vt16 = (_Float16*)((char*)d_ws + 2 * f16bytes);

    const int n4 = (int)(elems / 4);
    cvt_f32_to_f16<<<(n4 + 255) / 256, 256, 0, stream>>>(q, q16, n4);
    cvt_f32_to_f16<<<(n4 + 255) / 256, 256, 0, stream>>>(k, k16, n4);
    dim3 tg(SEQ / 32, DIM / 32, N_HEADS);
    transpose_v_f16<<<tg, 256, 0, stream>>>(v, vt16);

    attn_lnsm_f16<<<grid, block, lds_bytes, stream>>>(q16, k16, vt16, gamma, beta, out);
  } else {
    attn_lnsm_f32<<<grid, block, lds_bytes, stream>>>(q, k, v, gamma, beta, out);
  }
}
